// GNN_26482768347767
// MI455X (gfx1250) — matrix operand order repaired and checked by reading, not compiled, not run on hardware
//
#include <hip/hip_runtime.h>
#include <math.h>

typedef __attribute__((ext_vector_type(16))) _Float16 v16h;
typedef __attribute__((ext_vector_type(8)))  float    v8f;

#define DIM 256
#define FL  64

// ---------------- utility kernels ----------------

__global__ void k_zero(float* __restrict__ p, long long n) {
  long long i = (long long)blockIdx.x * blockDim.x + threadIdx.x;
  long long stride = (long long)gridDim.x * blockDim.x;
  for (; i < n; i += stride) p[i] = 0.0f;
}

__global__ void k_count_deg(const int* __restrict__ src, const int* __restrict__ dst,
                            float* __restrict__ dout, float* __restrict__ din, int E) {
  int e = blockIdx.x * blockDim.x + threadIdx.x;
  if (e >= E) return;
  atomicAdd(&dout[src[e]], 1.0f);
  atomicAdd(&din[dst[e]], 1.0f);
}

__global__ void k_finalize_deg(float* __restrict__ d, int N) {
  int i = blockIdx.x * blockDim.x + threadIdx.x;
  if (i >= N) return;
  d[i] = rsqrtf(fmaxf(d[i], 1.0f));
}

// bag-of-features: h0[n][f] = #slots with feat==f
__global__ void k_build_h0(const int* __restrict__ feat, float* __restrict__ h, int N) {
  int n = blockIdx.x * blockDim.x + threadIdx.x;
  if (n >= N) return;
  int f0 = feat[4*n+0], f1 = feat[4*n+1], f2 = feat[4*n+2], f3 = feat[4*n+3];
  float* row = h + (size_t)n * FL;
  for (int i = 0; i < FL; ++i) {
    row[i] = (float)((f0 == i) + (f1 == i) + (f2 == i) + (f3 == i));
  }
}

// agg[dst] += h[src] * deg_out^{-1/2}[src], 4 floats per thread
__global__ void k_aggregate(const float* __restrict__ h, const float* __restrict__ sout,
                            const int* __restrict__ src, const int* __restrict__ dst,
                            float* __restrict__ agg, int E, int D, int chunkShift) {
  long long t = (long long)blockIdx.x * blockDim.x + threadIdx.x;
  int chunks = D >> 2;
  if (t >= (long long)E * chunks) return;
  int e = (int)(t >> chunkShift);
  int c = ((int)t & (chunks - 1)) << 2;
  int s = src[e], d = dst[e];
  float sc = sout[s];
  const float4 v = *(const float4*)(h + (size_t)s * D + c);
  float* o = agg + (size_t)d * D + c;
  atomicAdd(o + 0, v.x * sc); atomicAdd(o + 1, v.y * sc);
  atomicAdd(o + 2, v.z * sc); atomicAdd(o + 3, v.w * sc);
}

// transpose+convert weights: Wt[n][k] (f16) = W[k][n] (f32)
__global__ void k_wt(const float* __restrict__ W, _Float16* __restrict__ Wt, int K, int Nout) {
  int t = blockIdx.x * blockDim.x + threadIdx.x;
  if (t >= K * Nout) return;
  int n = t / K, k = t - n * K;
  Wt[(size_t)n * K + k] = (_Float16)W[(size_t)k * Nout + n];
}

// ---------------- WMMA GEMM ----------------
// out[M,Dout] = act((A * sin_[row]) @ Wt^T + bias), A f32 row-major [M,K],
// Wt f16 [Dout,K] (pre-transposed weights).
// One wave per 16x64 output strip: A fragment converted once, 4 WMMAs per K-step.
// Requires Dout % 64 == 0 (Dout == 256 everywhere here).
__global__ void __launch_bounds__(256)
k_gemm_wmma(const float* __restrict__ A, const float* __restrict__ sin_,
            const _Float16* __restrict__ Wt, const float* __restrict__ bias,
            float* __restrict__ out, int M, int K, int Dout, int relu)
{
  int gt = blockIdx.x * blockDim.x + threadIdx.x;
  int wave = gt >> 5;
  int lane = threadIdx.x & 31;
  int stripsN = Dout >> 6;                       // 64-wide column strips
  int tilesM = (M + 15) >> 4;
  if (wave >= tilesM * stripsN) return;          // wave-uniform exit (EXEC all-1 below)
  int tm = wave / stripsN;
  int ts = wave - tm * stripsN;
  int m0 = tm << 4, n0 = ts << 6;

  // A fragment source: row m0+lane%16; two 8-half chunks at kbase, kbase+8
  int arow = m0 + (lane & 15);
  if (arow >= M) arow = M - 1;                   // clamp (duplicated rows never stored)
  float s = sin_ ? sin_[arow] : 1.0f;
  const float* aptr = A + (size_t)arow * K + ((lane < 16) ? 0 : 16);

  // B fragment source: column (n0 + sub*16 + lane%16) of W == row of Wt
  const _Float16* bptr = Wt + (size_t)(n0 + (lane & 15)) * K + ((lane < 16) ? 0 : 16);
  const size_t bsub = (size_t)16 * K;            // stride between 16-col subtiles

  v8f acc0 = {}, acc1 = {}, acc2 = {}, acc3 = {};
  for (int k0 = 0; k0 < K; k0 += 32) {
    const float4* a4  = (const float4*)(aptr + k0);
    const float4* a4b = (const float4*)(aptr + k0 + 8);
    float4 x0 = a4[0],  x1 = a4[1];
    float4 y0 = a4b[0], y1 = a4b[1];
    v16h af;
    af[0]  = (_Float16)(x0.x * s); af[1]  = (_Float16)(x0.y * s);
    af[2]  = (_Float16)(x0.z * s); af[3]  = (_Float16)(x0.w * s);
    af[4]  = (_Float16)(x1.x * s); af[5]  = (_Float16)(x1.y * s);
    af[6]  = (_Float16)(x1.z * s); af[7]  = (_Float16)(x1.w * s);
    af[8]  = (_Float16)(y0.x * s); af[9]  = (_Float16)(y0.y * s);
    af[10] = (_Float16)(y0.z * s); af[11] = (_Float16)(y0.w * s);
    af[12] = (_Float16)(y1.x * s); af[13] = (_Float16)(y1.y * s);
    af[14] = (_Float16)(y1.z * s); af[15] = (_Float16)(y1.w * s);

    v16h bf0 = *(const v16h*)(bptr + k0);
    v16h bf1 = *(const v16h*)(bptr + bsub + k0);
    v16h bf2 = *(const v16h*)(bptr + 2 * bsub + k0);
    v16h bf3 = *(const v16h*)(bptr + 3 * bsub + k0);

    acc0 = __builtin_amdgcn_wmma_f32_16x16x32_f16(false, af, false, bf0, (short)0, acc0, false, false);
    acc1 = __builtin_amdgcn_wmma_f32_16x16x32_f16(false, af, false, bf1, (short)0, acc1, false, false);
    acc2 = __builtin_amdgcn_wmma_f32_16x16x32_f16(false, af, false, bf2, (short)0, acc2, false, false);
    acc3 = __builtin_amdgcn_wmma_f32_16x16x32_f16(false, af, false, bf3, (short)0, acc3, false, false);
  }

  // C/D layout: VGPR j -> m = j + (lane>=16 ? 8 : 0), n = lane&15
  int mo = m0 + ((lane >> 4) << 3);
  v8f accs[4] = {acc0, acc1, acc2, acc3};
#pragma unroll
  for (int sub = 0; sub < 4; ++sub) {
    int nn = n0 + sub * 16 + (lane & 15);
    float bb = bias ? bias[nn] : 0.0f;
#pragma unroll
    for (int j = 0; j < 8; ++j) {
      int m = mo + j;
      if (m < M) {
        float v = accs[sub][j] + bb;
        if (relu) v = fmaxf(v, 0.0f);
        out[(size_t)m * Dout + nn] = v;
      }
    }
  }
}

// ---------------- factor / pooling / MLP tail ----------------

// sum of L2 row norms, wave per node
__global__ void k_norm_sum(const float* __restrict__ h, int N, float* __restrict__ accum) {
  int gt = blockIdx.x * blockDim.x + threadIdx.x;
  int wave = gt >> 5, lane = gt & 31;
  if (wave >= N) return;
  const float4* row = (const float4*)(h + (size_t)wave * DIM);
  float4 a = row[lane * 2], b = row[lane * 2 + 1];
  float s = a.x*a.x + a.y*a.y + a.z*a.z + a.w*a.w
          + b.x*b.x + b.y*b.y + b.z*b.z + b.w*b.w;
  for (int o = 16; o > 0; o >>= 1) s += __shfl_down(s, o, 32);
  if (lane == 0) atomicAdd(accum, sqrtf(s));
}

__global__ void k_factor(const float* __restrict__ accum, float* __restrict__ factor, float fN) {
  *factor = sqrtf((float)DIM) * fN / (*accum);
}

// embcat[gid[n]][colOff + d] += h[n][d] * factor
__global__ void k_pool(const float* __restrict__ h, const int* __restrict__ gid,
                       const float* __restrict__ factor, float* __restrict__ embcat,
                       int N, int colOff) {
  long long t = (long long)blockIdx.x * blockDim.x + threadIdx.x;
  if (t >= (long long)N * (DIM / 4)) return;
  int n = (int)(t >> 6);
  int c = ((int)t & 63) << 2;
  float f = *factor;
  float4 v = *(const float4*)(h + (size_t)n * DIM + c);
  float* o = embcat + (size_t)gid[n] * (2 * DIM) + colOff + c;
  atomicAdd(o + 0, v.x * f); atomicAdd(o + 1, v.y * f);
  atomicAdd(o + 2, v.z * f); atomicAdd(o + 3, v.w * f);
}

// out[g] = hidden[g,:256] . Wo + bo ; wave per graph
__global__ void k_final(const float* __restrict__ hidden, const float* __restrict__ Wo,
                        const float* __restrict__ bo, float* __restrict__ out, int G) {
  int gt = blockIdx.x * blockDim.x + threadIdx.x;
  int wave = gt >> 5, lane = gt & 31;
  if (wave >= G) return;
  const float4* hrow = (const float4*)(hidden + (size_t)wave * DIM);
  const float4* wv = (const float4*)Wo;
  float4 a = hrow[lane * 2], w = wv[lane * 2];
  float s = a.x*w.x + a.y*w.y + a.z*w.z + a.w*w.w;
  a = hrow[lane * 2 + 1]; w = wv[lane * 2 + 1];
  s += a.x*w.x + a.y*w.y + a.z*w.z + a.w*w.w;
  for (int o = 16; o > 0; o >>= 1) s += __shfl_down(s, o, 32);
  if (lane == 0) out[wave] = s + bo[0];
}

// ---------------- host orchestration ----------------

extern "C" void kernel_launch(void* const* d_in, const int* in_sizes, int n_in,
                              void* d_out, int out_size, void* d_ws, size_t ws_size,
                              hipStream_t stream) {
  (void)n_in; (void)ws_size;
  const int N = in_sizes[0] / 4;
  const int E = in_sizes[1];
  const int G = out_size;   // 2048

  const int* feat[2] = {(const int*)d_in[0], (const int*)d_in[4]};
  const int* src[2]  = {(const int*)d_in[1], (const int*)d_in[5]};
  const int* dst[2]  = {(const int*)d_in[2], (const int*)d_in[6]};
  const int* gid[2]  = {(const int*)d_in[3], (const int*)d_in[7]};
  const float* W0 = (const float*)d_in[8];  const float* b0 = (const float*)d_in[9];
  const float* W1 = (const float*)d_in[10]; const float* b1 = (const float*)d_in[11];
  const float* W2 = (const float*)d_in[12]; const float* b2 = (const float*)d_in[13];
  const float* Wh = (const float*)d_in[14]; const float* bh = (const float*)d_in[15];
  const float* Wo = (const float*)d_in[16]; const float* bo = (const float*)d_in[17];

  // workspace layout (256B aligned slices)
  size_t off = 0;
  auto take = [&](size_t bytes) { size_t o = off; off = (off + bytes + 255) & ~(size_t)255; return o; };
  char* base = (char*)d_ws;
  float* s_out  = (float*)(base + take((size_t)N * 4));
  float* s_in   = (float*)(base + take((size_t)N * 4));
  float* h      = (float*)(base + take((size_t)N * DIM * 4));
  float* agg    = (float*)(base + take((size_t)N * DIM * 4));
  float* embcat = (float*)(base + take((size_t)G * 2 * DIM * 4));
  float* hidden = (float*)(base + take((size_t)G * DIM * 4));
  float* scal   = (float*)(base + take(2 * 4));     // [0]=norm sum, [1]=factor
  _Float16* W0t = (_Float16*)(base + take((size_t)DIM * FL * 2));
  _Float16* W1t = (_Float16*)(base + take((size_t)DIM * DIM * 2));
  _Float16* W2t = (_Float16*)(base + take((size_t)DIM * DIM * 2));
  _Float16* Wht = (_Float16*)(base + take((size_t)DIM * 2 * DIM * 2));

  const int B = 256;
  auto nb = [&](long long n) { return (int)((n + B - 1) / B); };

  // weight transpose+f16 convert (tiny)
  k_wt<<<nb((long long)FL * DIM),     B, 0, stream>>>(W0, W0t, FL, DIM);
  k_wt<<<nb((long long)DIM * DIM),    B, 0, stream>>>(W1, W1t, DIM, DIM);
  k_wt<<<nb((long long)DIM * DIM),    B, 0, stream>>>(W2, W2t, DIM, DIM);
  k_wt<<<nb((long long)2 * DIM * DIM),B, 0, stream>>>(Wh, Wht, 2 * DIM, DIM);

  k_zero<<<1024, B, 0, stream>>>(embcat, (long long)G * 2 * DIM);
  k_zero<<<1, 32, 0, stream>>>(scal, 2);

  const int gemmWaves = ((N + 15) / 16) * (DIM / 64);   // 16x64 strips

  for (int g = 0; g < 2; ++g) {
    // degree normalizers
    k_zero<<<1024, B, 0, stream>>>(s_out, N);
    k_zero<<<1024, B, 0, stream>>>(s_in, N);
    k_count_deg<<<nb(E), B, 0, stream>>>(src[g], dst[g], s_out, s_in, E);
    k_finalize_deg<<<nb(N), B, 0, stream>>>(s_out, N);
    k_finalize_deg<<<nb(N), B, 0, stream>>>(s_in, N);

    // embedding
    k_build_h0<<<nb(N), B, 0, stream>>>(feat[g], h, N);

    // layer 0 (K = 64)
    k_zero<<<2048, B, 0, stream>>>(agg, (long long)N * FL);
    k_aggregate<<<nb((long long)E * (FL / 4)), B, 0, stream>>>(h, s_out, src[g], dst[g], agg, E, FL, 4);
    k_gemm_wmma<<<nb((long long)gemmWaves * 32), B, 0, stream>>>(agg, s_in, W0t, b0, h, N, FL, DIM, 1);

    // layers 1,2 (K = 256)
    const _Float16* Wts[2] = {W1t, W2t};
    const float* bs[2] = {b1, b2};
    for (int l = 0; l < 2; ++l) {
      k_zero<<<4096, B, 0, stream>>>(agg, (long long)N * DIM);
      k_aggregate<<<nb((long long)E * (DIM / 4)), B, 0, stream>>>(h, s_out, src[g], dst[g], agg, E, DIM, 6);
      k_gemm_wmma<<<nb((long long)gemmWaves * 32), B, 0, stream>>>(agg, s_in, Wts[l], bs[l], h, N, DIM, DIM,
                                                                   (l == 0) ? 1 : 0);
    }

    // factor from graph 1 only, reused for graph 2
    if (g == 0) {
      k_norm_sum<<<nb((long long)N * 32), B, 0, stream>>>(h, N, &scal[0]);
      k_factor<<<1, 1, 0, stream>>>(&scal[0], &scal[1], (float)N);
    }
    k_pool<<<nb((long long)N * (DIM / 4)), B, 0, stream>>>(h, gid[g], &scal[1], embcat, N, (g == 0) ? 0 : DIM);
  }

  // MLP hidden = relu(embcat @ Wh + bh), K = 512
  const int mlpWaves = ((G + 15) / 16) * (DIM / 64);
  k_gemm_wmma<<<nb((long long)mlpWaves * 32), B, 0, stream>>>(embcat, nullptr, Wht, bh, hidden,
                                                              G, 2 * DIM, DIM, 1);
  // out = hidden @ Wo + bo
  k_final<<<nb((long long)G * 32), B, 0, stream>>>(hidden, Wo, bo, (float*)d_out, G);
}
